// MultiHeadAttention_84920093376856
// MI455X (gfx1250) — matrix operand order repaired and checked by reading, not compiled, not run on hardware
//
#include <hip/hip_runtime.h>

typedef __attribute__((ext_vector_type(16))) _Float16 v16h;
typedef __attribute__((ext_vector_type(8)))  _Float16 v8h;
typedef __attribute__((ext_vector_type(8)))  float    v8f;
typedef __attribute__((ext_vector_type(4)))  float    v4f;

#define NB 512      // batch
#define NT 128      // seq len
#define NC 768      // embed
#define NH 12       // heads
#define ND 64       // head size

union F16Frag { v16h v; v8h h[2]; };

// ---------------------------------------------------------------------------
// Generic GEMM:  out[m][n] = sum_k A[m][k] * W[n][k]   (W row-major [N,K])
// A_F16: A is f16 (attn buffer) vs f32 (x).  EPI: 0 = scatter qkv f16, 1 = bias+f32 out
// Block: 128x128 tile, BK=32, 256 threads = 8 waves (4 along M x 2 along N),
// each wave owns a 32x64 sub-tile = 2x4 WMMA accumulators.
// Double-buffered LDS staging; grid is (N-tiles, M-tiles) so the 18/6 blocks
// sharing one A row-panel dispatch back-to-back and hit it in L2.
// ---------------------------------------------------------------------------
template <int A_F16, int EPI>
__global__ __launch_bounds__(256)
void gemm_f16wmma(const void* __restrict__ Aptr, const float* __restrict__ W,
                  const float* __restrict__ bias,
                  _Float16* __restrict__ q_out, _Float16* __restrict__ k_out,
                  _Float16* __restrict__ v_out, float* __restrict__ out,
                  int M, int N, int K)
{
    const int LS = 40; // LDS row stride (halfs): 80B rows, 16B aligned, bank-skewed
    __shared__ _Float16 ldsA[2][128 * LS];
    __shared__ _Float16 ldsB[2][128 * LS];

    const int tid  = threadIdx.x;
    const int lane = tid & 31;
    const int wave = tid >> 5;
    const int wm = wave & 3;       // wave M index (0..3) -> 32 rows each
    const int wn = wave >> 2;      // wave N index (0..1) -> 64 cols each
    const int blockN = blockIdx.x * 128;   // N fastest-varying -> A-panel L2 reuse
    const int blockM = blockIdx.y * 128;

    const int lanem = lane & 15;
    const int laneh = lane >> 4;

    v8f acc[2][4];
#pragma unroll
    for (int mi = 0; mi < 2; ++mi)
#pragma unroll
        for (int ni = 0; ni < 4; ++ni) acc[mi][ni] = (v8f){};

    const int lrow = tid >> 1;          // 0..127
    const int lcol = (tid & 1) * 16;    // 0 or 16

    const _Float16* Af16 = (const _Float16*)Aptr;
    const float*    Af32 = (const float*)Aptr;

    // stage-load helpers: global (f32 or f16) -> two v8h register chunks
    auto loadA = [&](int k0, v8h& h0, v8h& h1) {
        if (A_F16) {
            const _Float16* s = Af16 + (size_t)(blockM + lrow) * K + k0 + lcol;
            h0 = *(const v8h*)(s);
            h1 = *(const v8h*)(s + 8);
        } else {
            const float* s = Af32 + (size_t)(blockM + lrow) * K + k0 + lcol;
            v4f f0 = *(const v4f*)(s + 0);
            v4f f1 = *(const v4f*)(s + 4);
            v4f f2 = *(const v4f*)(s + 8);
            v4f f3 = *(const v4f*)(s + 12);
#pragma unroll
            for (int i = 0; i < 4; ++i) {
                h0[i]     = (_Float16)f0[i];
                h0[4 + i] = (_Float16)f1[i];
                h1[i]     = (_Float16)f2[i];
                h1[4 + i] = (_Float16)f3[i];
            }
        }
    };
    auto loadW = [&](int k0, v8h& h0, v8h& h1) {
        const float* s = W + (size_t)(blockN + lrow) * K + k0 + lcol;
        v4f f0 = *(const v4f*)(s + 0);
        v4f f1 = *(const v4f*)(s + 4);
        v4f f2 = *(const v4f*)(s + 8);
        v4f f3 = *(const v4f*)(s + 12);
#pragma unroll
        for (int i = 0; i < 4; ++i) {
            h0[i]     = (_Float16)f0[i];
            h0[4 + i] = (_Float16)f1[i];
            h1[i]     = (_Float16)f2[i];
            h1[4 + i] = (_Float16)f3[i];
        }
    };

    // ---- prologue: stage tile 0 into buffer 0 ----
    v8h a0, a1, b0, b1;
    loadA(0, a0, a1);
    loadW(0, b0, b1);
    *(v8h*)&ldsA[0][lrow * LS + lcol]     = a0;
    *(v8h*)&ldsA[0][lrow * LS + lcol + 8] = a1;
    *(v8h*)&ldsB[0][lrow * LS + lcol]     = b0;
    *(v8h*)&ldsB[0][lrow * LS + lcol + 8] = b1;
    __syncthreads();

    const int KT = K >> 5;   // number of 32-wide K tiles
    for (int kt = 0; kt < KT; ++kt) {
        const int cur = kt & 1;
        const bool hasNext = (kt + 1 < KT);

        // issue next tile's global loads before compute (latency hides under WMMA)
        if (hasNext) {
            loadA((kt + 1) * 32, a0, a1);
            loadW((kt + 1) * 32, b0, b1);
        }
        // pull the tile after that toward L2 (global_prefetch_b8)
        if (kt + 2 < KT) {
            const int kp = (kt + 2) * 32;
            if (A_F16)
                __builtin_prefetch((const void*)(Af16 + (size_t)(blockM + lrow) * K + kp + lcol), 0, 0);
            else
                __builtin_prefetch((const void*)(Af32 + (size_t)(blockM + lrow) * K + kp + lcol), 0, 0);
            __builtin_prefetch((const void*)(W + (size_t)(blockN + lrow) * K + kp + lcol), 0, 0);
        }

        // ---- fragments + 8 WMMAs from current buffer ----
        F16Frag af[2];
#pragma unroll
        for (int mi = 0; mi < 2; ++mi) {
            const int ar = (32 * wm + 16 * mi + lanem) * LS;
            af[mi].h[0] = *(const v8h*)&ldsA[cur][ar + 8 * laneh];
            af[mi].h[1] = *(const v8h*)&ldsA[cur][ar + 16 + 8 * laneh];
        }
        F16Frag bf[4];
#pragma unroll
        for (int ni = 0; ni < 4; ++ni) {
            const int br = (64 * wn + 16 * ni + lanem) * LS + 8 * laneh;
            bf[ni].h[0] = *(const v8h*)&ldsB[cur][br];
            bf[ni].h[1] = *(const v8h*)&ldsB[cur][br + 16];
        }
#pragma unroll
        for (int mi = 0; mi < 2; ++mi)
#pragma unroll
            for (int ni = 0; ni < 4; ++ni)
                acc[mi][ni] = __builtin_amdgcn_wmma_f32_16x16x32_f16(
                    false, af[mi].v, false, bf[ni].v, (short)0, acc[mi][ni],
                    false, false);

        // ---- stage next tile into the alternate buffer ----
        if (hasNext) {
            *(v8h*)&ldsA[cur ^ 1][lrow * LS + lcol]     = a0;
            *(v8h*)&ldsA[cur ^ 1][lrow * LS + lcol + 8] = a1;
            *(v8h*)&ldsB[cur ^ 1][lrow * LS + lcol]     = b0;
            *(v8h*)&ldsB[cur ^ 1][lrow * LS + lcol + 8] = b1;
        }
        __syncthreads();
    }

    // ---- epilogue ----
    const int m0 = blockM + 32 * wm;
    const int n0 = blockN + 64 * wn;
#pragma unroll
    for (int mi = 0; mi < 2; ++mi) {
#pragma unroll
        for (int ni = 0; ni < 4; ++ni) {
#pragma unroll
            for (int r = 0; r < 8; ++r) {
                const int mg = m0 + 16 * mi + r + 8 * laneh;
                const int ng = n0 + 16 * ni + lanem;
                const float val = acc[mi][ni][r];
                if (EPI == 0) {
                    // scatter q/k/v as f16 in [B,H,T,D] layout
                    const int sec = ng / NC;            // 0=q 1=k 2=v
                    const int within = ng - sec * NC;
                    const int hh = within >> 6;
                    const int d  = within & 63;
                    const int b  = mg >> 7;             // T = 128
                    const int t  = mg & 127;
                    const size_t off =
                        ((size_t)(b * NH + hh) * NT + t) * ND + d;
                    _Float16* dst = (sec == 0) ? q_out : (sec == 1) ? k_out : v_out;
                    dst[off] = (_Float16)val;
                } else {
                    out[(size_t)mg * N + ng] = val + bias[ng];
                }
            }
        }
    }
}

// ---------------------------------------------------------------------------
// Attention: one block per (b,h). 8 waves; wave w owns rows [16w,16w+16).
// S = Q K^T in registers (16 WMMAs/wave), mask+softmax in registers via
// half-wave shuffles, P spilled f16 to per-wave LDS strip, O = P V (16 WMMAs).
// ---------------------------------------------------------------------------
__global__ __launch_bounds__(256)
void attn_f16wmma(const _Float16* __restrict__ qf, const _Float16* __restrict__ kf,
                  const _Float16* __restrict__ vf, _Float16* __restrict__ attn_out)
{
    const int bh = blockIdx.x;           // b*NH + h
    const int b  = bh / NH;
    const int h  = bh - b * NH;
    const _Float16* Q  = qf + (size_t)bh * NT * ND;
    const _Float16* Km = kf + (size_t)bh * NT * ND;
    const _Float16* V  = vf + (size_t)bh * NT * ND;

    __shared__ _Float16 Vt[ND][136];        // Vt[d][s] = V[s][d]
    __shared__ _Float16 Plds[8][16][136];   // per-wave P strip (16 x 128)

    const int tid   = threadIdx.x;
    const int lane  = tid & 31;
    const int wave  = tid >> 5;
    const int lanem = lane & 15;
    const int laneh = lane >> 4;
    const int mBase = 16 * wave;

    // transpose V into LDS
    for (int i = tid; i < NT * ND; i += 256) {
        const int s = i >> 6;
        const int d = i & 63;
        Vt[d][s] = V[i];
    }
    __syncthreads();

    // ---- S = Q K^T : rows [mBase, mBase+16), all 128 cols, K-dim = 64 ----
    F16Frag qa[2];
    {
        const _Float16* qrow = Q + (size_t)(mBase + lanem) * ND;
#pragma unroll
        for (int ks = 0; ks < 2; ++ks) {
            qa[ks].h[0] = *(const v8h*)(qrow + 32 * ks + 8 * laneh);
            qa[ks].h[1] = *(const v8h*)(qrow + 32 * ks + 16 + 8 * laneh);
        }
    }
    v8f sacc[8];
#pragma unroll
    for (int t = 0; t < 8; ++t) {
        sacc[t] = (v8f){};
        const _Float16* krow = Km + (size_t)(16 * t + lanem) * ND;
#pragma unroll
        for (int ks = 0; ks < 2; ++ks) {
            F16Frag kb;
            kb.h[0] = *(const v8h*)(krow + 32 * ks + 8 * laneh);
            kb.h[1] = *(const v8h*)(krow + 32 * ks + 16 + 8 * laneh);
            sacc[t] = __builtin_amdgcn_wmma_f32_16x16x32_f16(
                false, qa[ks].v, false, kb.v, (short)0, sacc[t], false, false);
        }
    }

    // ---- causal mask + softmax(s * 1/sqrt(D)), row-wise in registers ----
    const float scale = 0.125f;           // 1/sqrt(64)
#pragma unroll
    for (int r = 0; r < 8; ++r) {
        const int m = mBase + r + 8 * laneh;   // global row (this half-wave's row)
        float vals[8];
        float rowmax = -3.0e38f;
#pragma unroll
        for (int t = 0; t < 8; ++t) {
            const int n = 16 * t + lanem;
            float sv = sacc[t][r] * scale;
            if (n > m) sv = -3.0e38f;
            vals[t] = sv;
            rowmax = fmaxf(rowmax, sv);
        }
#pragma unroll
        for (int off = 1; off < 16; off <<= 1)
            rowmax = fmaxf(rowmax, __shfl_xor(rowmax, off, 32));
        float rowsum = 0.0f;
#pragma unroll
        for (int t = 0; t < 8; ++t) {
            vals[t] = __expf(vals[t] - rowmax);
            rowsum += vals[t];
        }
#pragma unroll
        for (int off = 1; off < 16; off <<= 1)
            rowsum += __shfl_xor(rowsum, off, 32);
        const float inv = 1.0f / rowsum;
#pragma unroll
        for (int t = 0; t < 8; ++t)
            Plds[wave][r + 8 * laneh][16 * t + lanem] = (_Float16)(vals[t] * inv);
    }

    // ---- O = P V : [16 x 64], K-dim = 128 (4 k-steps) ----
    v8f oacc[4];
#pragma unroll
    for (int dn = 0; dn < 4; ++dn) oacc[dn] = (v8f){};
#pragma unroll
    for (int ks = 0; ks < 4; ++ks) {
        F16Frag pa;
        pa.h[0] = *(const v8h*)&Plds[wave][lanem][32 * ks + 8 * laneh];
        pa.h[1] = *(const v8h*)&Plds[wave][lanem][32 * ks + 16 + 8 * laneh];
#pragma unroll
        for (int dn = 0; dn < 4; ++dn) {
            F16Frag vb;
            vb.h[0] = *(const v8h*)&Vt[16 * dn + lanem][32 * ks + 8 * laneh];
            vb.h[1] = *(const v8h*)&Vt[16 * dn + lanem][32 * ks + 16 + 8 * laneh];
            oacc[dn] = __builtin_amdgcn_wmma_f32_16x16x32_f16(
                false, pa.v, false, vb.v, (short)0, oacc[dn], false, false);
        }
    }

    // ---- write O (f16) into [B*T, C] for the proj GEMM ----
#pragma unroll
    for (int dn = 0; dn < 4; ++dn) {
#pragma unroll
        for (int r = 0; r < 8; ++r) {
            const int t = mBase + r + 8 * laneh;
            const int c = h * ND + 16 * dn + lanem;
            attn_out[(size_t)(b * NT + t) * NC + c] = (_Float16)oacc[dn][r];
        }
    }
}

// ---------------------------------------------------------------------------
extern "C" void kernel_launch(void* const* d_in, const int* in_sizes, int n_in,
                              void* d_out, int out_size, void* d_ws, size_t ws_size,
                              hipStream_t stream) {
    const float* x      = (const float*)d_in[0];   // [B,T,C]
    const float* w_qkv  = (const float*)d_in[1];   // [3C, C]
    const float* w_proj = (const float*)d_in[2];   // [C, C]
    const float* b_proj = (const float*)d_in[3];   // [C]
    float* out = (float*)d_out;                    // [B,T,C]

    const size_t QKV_ELEMS = (size_t)NB * NH * NT * ND;   // 50,331,648
    _Float16* qf   = (_Float16*)d_ws;
    _Float16* kf   = qf + QKV_ELEMS;
    _Float16* vf   = kf + QKV_ELEMS;
    _Float16* attn = vf + QKV_ELEMS;                      // [B*T, C] f16

    const int M = NB * NT;                                // 65536

    // 1) qkv = x @ w_qkv^T, scattered to q/k/v f16 [B,H,T,D]
    {
        dim3 grid((3 * NC) / 128, M / 128);               // 18 x 512 (N fastest)
        gemm_f16wmma<0, 0><<<grid, 256, 0, stream>>>(
            x, w_qkv, nullptr, qf, kf, vf, nullptr, M, 3 * NC, NC);
    }
    // 2) causal attention per (b,h)
    attn_f16wmma<<<NB * NH, 256, 0, stream>>>(qf, kf, vf, attn);
    // 3) out = attn @ w_proj^T + b_proj
    {
        dim3 grid(NC / 128, M / 128);                     // 6 x 512 (N fastest)
        gemm_f16wmma<1, 1><<<grid, 256, 0, stream>>>(
            attn, w_proj, b_proj, nullptr, nullptr, nullptr, out, M, NC, NC);
    }
}
